// MLPPredictor_17798344474629
// MI455X (gfx1250) — matrix operand order repaired and checked by reading, not compiled, not run on hardware
//
#include <hip/hip_runtime.h>
#include <hip/hip_bf16.h>

typedef __attribute__((ext_vector_type(16))) _Float16 v16h;
typedef __attribute__((ext_vector_type(8)))  float    v8f;
typedef __attribute__((ext_vector_type(8)))  unsigned int v8u;

#define D_IN 64
#define HID  256
#define KIN  128     // 2*D
#define OUTF 2
#define OUTP 16      // padded out features for the 16x16 WMMA tile

// ---- workspace / LDS weight-block layout (bytes) ----
#define W1T_OFF  0
#define W1T_BYTES (HID*KIN*2)            // 65536   W1^T [256 x 128] f16
#define W2T_OFF  (W1T_OFF + W1T_BYTES)   // 65536
#define W2T_BYTES (HID*HID*2)            // 131072  W2^T [256 x 256] f16
#define W3T_OFF  (W2T_OFF + W2T_BYTES)   // 196608
#define W3T_BYTES (OUTP*HID*2)           // 8192    W3^T padded [16 x 256] f16
#define B1_OFF   (W3T_OFF + W3T_BYTES)   // 204800  b1 f32[256]
#define B2_OFF   (B1_OFF + HID*4)        // 205824  b2 f32[256]
#define B3_OFF   (B2_OFF + HID*4)        // 206848  b3 padded f32[16]
#define WB_BYTES (B3_OFF + OUTP*4)       // 206912  (multiple of 16)
#define H16_OFF  WB_BYTES                // h in f16 follows in workspace

// ---------------- prep kernels ----------------
__global__ void k_cvt_h(const float* __restrict__ h, _Float16* __restrict__ h16, int n) {
  int i = blockIdx.x * 256 + threadIdx.x;
  if (i < n) h16[i] = (_Float16)h[i];
}

__global__ void k_prep_w(const float* __restrict__ W1, const float* __restrict__ b1,
                         const float* __restrict__ W2, const float* __restrict__ b2,
                         const float* __restrict__ W3, const float* __restrict__ b3,
                         char* __restrict__ wb) {
  int i = blockIdx.x * 256 + threadIdx.x;
  _Float16* w1t = (_Float16*)(wb + W1T_OFF);
  _Float16* w2t = (_Float16*)(wb + W2T_OFF);
  _Float16* w3t = (_Float16*)(wb + W3T_OFF);
  float* ob1 = (float*)(wb + B1_OFF);
  float* ob2 = (float*)(wb + B2_OFF);
  float* ob3 = (float*)(wb + B3_OFF);
  if (i < HID*KIN)  { int n = i / KIN, k = i % KIN; w1t[i] = (_Float16)W1[k*HID + n]; }
  if (i < HID*HID)  { int n = i / HID, k = i % HID; w2t[i] = (_Float16)W2[k*HID + n]; }
  if (i < OUTP*HID) { int n = i / HID, k = i % HID;
                      w3t[i] = (n < OUTF) ? (_Float16)W3[k*OUTF + n] : (_Float16)0.f; }
  if (i < HID)  { ob1[i] = b1[i]; ob2[i] = b2[i]; }
  if (i < OUTP) { ob3[i] = (i < OUTF) ? b3[i] : 0.f; }
}

// ---------------- device helpers ----------------
__device__ __forceinline__ v8f wmma16x16x32(v8u a, v8u b, v8f c) {
  return __builtin_amdgcn_wmma_f32_16x16x32_f16(
      false, __builtin_bit_cast(v16h, a),
      false, __builtin_bit_cast(v16h, b),
      (short)0, c, false, false);
}

// 32 contiguous bytes (16 halfs) -> one B fragment lane-chunk (K base..base+15)
__device__ __forceinline__ v8u load32g(const char* p) {
  uint4 a = *(const uint4*)p;
  uint4 b = *(const uint4*)(p + 16);
  v8u r = {a.x, a.y, a.z, a.w, b.x, b.y, b.z, b.w};
  return r;
}

// One 16-outF x 16-edge output tile: acc initialized from bias, KT k-steps of WMMA.
// Weights are W^T row-major f16 in LDS; A-fragment pattern: lane r=L%16 reads row
// (16j+r); v0..3 = halfs [32t+16hi, +8), v4..7 = halfs [32t+16hi+8, +8).
template<int KT>
__device__ __forceinline__ v8f mtile(const char* w, int rowBytes, const float* bias,
                                     int j, int r, int hi, const v8u* B) {
  const float4* bp = (const float4*)(bias + j*16 + hi*8);
  float4 ba = bp[0], bb = bp[1];
  v8f acc = {ba.x, ba.y, ba.z, ba.w, bb.x, bb.y, bb.z, bb.w};
  const char* row = w + (j*16 + r) * rowBytes + hi*32;
#pragma unroll
  for (int t = 0; t < KT; ++t) {
    uint4 a0 = *(const uint4*)(row + t*64);
    uint4 a1 = *(const uint4*)(row + t*64 + 16);
    v8u a = {a0.x, a0.y, a0.z, a0.w, a1.x, a1.y, a1.z, a1.w};
    acc = wmma16x16x32(a, B[t], acc);
  }
  return acc;
}

// relu(a),relu(b) -> packed 2xf16 in one dword (v_med3 clamp + v_cvt_pk_f16_f32)
__device__ __forceinline__ unsigned pack_relu2(float a, float b) {
  float ra = __builtin_amdgcn_fmed3f(a, 0.f, __builtin_inff());
  float rb = __builtin_amdgcn_fmed3f(b, 0.f, __builtin_inff());
  auto p = __builtin_amdgcn_cvt_pkrtz(ra, rb);   // __fp16 ext_vector(2)
  return __builtin_bit_cast(unsigned, p);
}

__device__ __forceinline__ void pack_relu(const v8f& acc, unsigned (&P)[4]) {
#pragma unroll
  for (int i = 0; i < 4; ++i) P[i] = pack_relu2(acc[2*i], acc[2*i+1]);
}

// D-tile pair (rows 0..15 and 16..31 of Y^T) -> next-layer B fragment (K=32).
// B lane-lo holds K 0..15, lane-hi K 16..31; D lane-lo holds rows 0..7, hi 8..15,
// so swap halves with shfl_xor(16).
__device__ __forceinline__ v8u combine(const unsigned (&P0)[4], const unsigned (&P1)[4], bool lolane) {
  v8u b;
#pragma unroll
  for (int i = 0; i < 4; ++i) {
    unsigned q0 = __shfl_xor(P0[i], 16, 32);
    unsigned q1 = __shfl_xor(P1[i], 16, 32);
    b[i]     = lolane ? P0[i] : q1;
    b[4 + i] = lolane ? q0    : P1[i];
  }
  return b;
}

// ---------------- main kernel ----------------
__global__ void __launch_bounds__(256)
edge_mlp_kernel(const int* __restrict__ src, const int* __restrict__ dst,
                const _Float16* __restrict__ h16, const char* __restrict__ wblock,
                float* __restrict__ out, int E, int numTiles) {
  __shared__ __align__(16) char smem[WB_BYTES];   // 207 KB: all weights + biases
  {
    const uint4* g = (const uint4*)wblock;
    uint4* s = (uint4*)smem;
    for (int i = threadIdx.x; i < WB_BYTES / 16; i += 256) s[i] = g[i];
  }
  __syncthreads();

  const int lane = threadIdx.x & 31;
  const int wave = threadIdx.x >> 5;
  const int r    = lane & 15;       // edge column within tile
  const int hi   = lane >> 4;       // lane half
  const bool lolane = (hi == 0);

  for (int tile = blockIdx.x * 8 + wave; tile < numTiles; tile += gridDim.x * 8) {
    // Opaque zero: makes LDS weight addresses loop-variant so the compiler
    // cannot hoist (and then spill) the weight fragments out of the loop.
    int zoff;
    asm volatile("s_mov_b32 %0, 0" : "=s"(zoff));
    const char*  W1t = smem + W1T_OFF + zoff;
    const char*  W2t = smem + W2T_OFF + zoff;
    const char*  W3t = smem + W3T_OFF + zoff;
    const float* b1  = (const float*)(smem + B1_OFF + zoff);
    const float* b2  = (const float*)(smem + B2_OFF + zoff);
    const float* b3  = (const float*)(smem + B3_OFF + zoff);

    const int e  = tile * 16 + r;
    const int ec = e < E ? e : (E - 1);
    const int sn = src[ec];
    const int dn = dst[ec];
    const char* srow = (const char*)(h16 + (size_t)sn * D_IN);
    const char* drow = (const char*)(h16 + (size_t)dn * D_IN);

    // Layer-1 B fragments: X1^T [K=128 features, N=16 edges], 4 k-tiles of 32.
    v8u B1f[4];
    B1f[0] = load32g(srow +      hi * 32);   // src features  0..31
    B1f[1] = load32g(srow + 64 + hi * 32);   // src features 32..63
    B1f[2] = load32g(drow +      hi * 32);   // dst features  0..31
    B1f[3] = load32g(drow + 64 + hi * 32);   // dst features 32..63

    unsigned P0[4], P1[4];

    // Layer 1: Y1^T = W1^T (16x32 A tiles) x X1^T -> B fragments for layer 2
    v8u B2f[8];
#pragma unroll
    for (int u = 0; u < 8; ++u) {
      v8f a0 = mtile<4>(W1t, KIN * 2, b1, 2*u,     r, hi, B1f);
      pack_relu(a0, P0);
      v8f a1 = mtile<4>(W1t, KIN * 2, b1, 2*u + 1, r, hi, B1f);
      pack_relu(a1, P1);
      B2f[u] = combine(P0, P1, lolane);
    }

    // Layer 2: Y2^T = W2^T x Y1^T -> B fragments for layer 3
    v8u B3f[8];
#pragma unroll
    for (int u = 0; u < 8; ++u) {
      v8f a0 = mtile<8>(W2t, HID * 2, b2, 2*u,     r, hi, B2f);
      pack_relu(a0, P0);
      v8f a1 = mtile<8>(W2t, HID * 2, b2, 2*u + 1, r, hi, B2f);
      pack_relu(a1, P1);
      B3f[u] = combine(P0, P1, lolane);
    }

    // Layer 3: single padded 16x16 output tile; rows 0,1 are the real scores.
    v8f o = mtile<8>(W3t, HID * 2, b3, 0, r, hi, B3f);
    if (lolane && e < E) {
      float2 v; v.x = o[0]; v.y = o[1];
      reinterpret_cast<float2*>(out)[e] = v;
    }
  }
}

// ---------------- launch ----------------
extern "C" void kernel_launch(void* const* d_in, const int* in_sizes, int n_in,
                              void* d_out, int out_size, void* d_ws, size_t ws_size,
                              hipStream_t stream) {
  const float* h  = (const float*)d_in[0];
  const int*   src = (const int*)d_in[1];
  const int*   dst = (const int*)d_in[2];
  const float* W1 = (const float*)d_in[3];
  const float* b1 = (const float*)d_in[4];
  const float* W2 = (const float*)d_in[5];
  const float* b2 = (const float*)d_in[6];
  const float* W3 = (const float*)d_in[7];
  const float* b3 = (const float*)d_in[8];
  const int nh = in_sizes[0];   // N_NODES * 64
  const int E  = in_sizes[1];

  char* wb = (char*)d_ws;
  _Float16* h16 = (_Float16*)(wb + H16_OFF);
  float* out = (float*)d_out;

  k_cvt_h <<<(nh + 255) / 256, 256, 0, stream>>>(h, h16, nh);
  k_prep_w<<<(HID * HID + 255) / 256, 256, 0, stream>>>(W1, b1, W2, b2, W3, b3, wb);

  int numTiles = (E + 15) / 16;
  int blocks = (numTiles + 7) / 8;
  if (blocks > 640) blocks = 640;
  edge_mlp_kernel<<<blocks, 256, 0, stream>>>(src, dst, h16, wb, out, E, numTiles);
}
